// MHA_45964740002076
// MI455X (gfx1250) — matrix operand order repaired and checked by reading, not compiled, not run on hardware
//
#include <hip/hip_runtime.h>

// ---------------- types ----------------
typedef __bf16 v8bf  __attribute__((ext_vector_type(8)));
typedef __bf16 v16bf __attribute__((ext_vector_type(16)));
typedef float  v8f   __attribute__((ext_vector_type(8)));
typedef unsigned int u32x4 __attribute__((ext_vector_type(4)));
typedef int          i32x4 __attribute__((ext_vector_type(4)));
typedef int          i32x8 __attribute__((ext_vector_type(8)));

#define DMODEL 768
#define NSEQ   4096
#define NHEAD  12
#define DHEAD  64
#define D3     2304

// fp32 -> bf16 round-to-nearest-even
__device__ __forceinline__ unsigned short f2bf(float f) {
  unsigned int u = __builtin_bit_cast(unsigned int, f);
  unsigned int r = u + 0x7FFFu + ((u >> 16) & 1u);
  return (unsigned short)(r >> 16);
}

__device__ __forceinline__ v16bf cat8(v8bf lo, v8bf hi) {
  return __builtin_shufflevector(lo, hi, 0,1,2,3,4,5,6,7,8,9,10,11,12,13,14,15);
}

// A-fragment (16x32 bf16, M=row per lane): rowPtr points at (row, ks).
__device__ __forceinline__ v16bf ldA32(const unsigned short* rowPtr, int g) {
  const v8bf* p = (const v8bf*)rowPtr;
  return cat8(p[2 * g], p[2 * g + 1]);
}

// B-fragment (32x16 bf16, N=col per lane): ptr points at (col, ks + g*16).
__device__ __forceinline__ v16bf ldB32(const unsigned short* ptr) {
  const v8bf* p = (const v8bf*)ptr;
  return cat8(p[0], p[1]);
}

__device__ __forceinline__ v8f wmma_bf16(v16bf a, v16bf b, v8f c) {
  return __builtin_amdgcn_wmma_f32_16x16x32_bf16(
      false, a, false, b, (short)0, c, false, false);
}

// 16B async copy global -> LDS (per lane), tracked by ASYNCcnt.
__device__ __forceinline__ void async_copy16(unsigned lds_addr, const void* base, unsigned voff) {
  asm volatile("global_load_async_to_lds_b128 %0, %1, %2"
               :: "v"(lds_addr), "v"(voff), "s"(base)
               : "memory");
}

// TDM: move a 2D tile (tile0 x tile1 elements of 2B, row stride stride0 elements)
// from global to LDS with a single tensor_load_to_lds (TENSORcnt-tracked).
// Descriptor fields per CDNA5 ISA 8.3/8.4.
__device__ __forceinline__ void tdm_load_2d_bf16(unsigned lds_addr, const void* gsrc,
                                                 unsigned tensor0, unsigned tensor1,
                                                 unsigned stride0, unsigned tile0,
                                                 unsigned tile1) {
  unsigned long long ga = (unsigned long long)(size_t)gsrc;
  u32x4 g0;
  g0[0] = 1u;                                            // count=1 (valid user D#)
  g0[1] = lds_addr;                                      // lds_addr [63:32]
  g0[2] = (unsigned)(ga & 0xFFFFFFFFu);                  // global_addr [95:64]
  g0[3] = (unsigned)((ga >> 32) & 0x01FFFFFFu) | (2u << 30);  // addr[56:32] | type=2
  i32x8 g1;
  g1[0] = (int)(1u << 16);                               // data_size=1 (2 bytes)
  g1[1] = (int)((tensor0 & 0xFFFFu) << 16);              // tensor_dim0 lo16 @ [63:48]
  g1[2] = (int)((tensor0 >> 16) | ((tensor1 & 0xFFFFu) << 16)); // dim0 hi | dim1 lo
  g1[3] = (int)((tensor1 >> 16) | (tile0 << 16));        // dim1 hi | tile_dim0
  g1[4] = (int)(tile1);                                  // tile_dim1 (tile_dim2=0)
  g1[5] = (int)(stride0);                                // tensor_dim0_stride lo32
  g1[6] = 0;                                             // stride0 hi | stride1 lo
  g1[7] = 0;                                             // stride1 hi
  i32x4 gz = {0, 0, 0, 0};
#if __has_include(<hip/amd_detail/amd_gfx1250_TDM.h>)
  i32x8 z8 = {0, 0, 0, 0, 0, 0, 0, 0};
  __builtin_amdgcn_tensor_load_to_lds(g0, g1, gz, gz, z8, 0);
#else
  __builtin_amdgcn_tensor_load_to_lds(g0, g1, gz, gz, 0);
#endif
}

// ---------------- conversion kernels ----------------
__global__ __launch_bounds__(256) void cvt_f32_bf16(const float* __restrict__ src,
                                                    unsigned short* __restrict__ dst, int n) {
  int i = blockIdx.x * 256 + threadIdx.x;
  if (i < n) dst[i] = f2bf(src[i]);
}

// src[rows][cols] fp32 -> dst[cols][rows] bf16
__global__ __launch_bounds__(256) void transpose_cvt(const float* __restrict__ src,
                                                     unsigned short* __restrict__ dst,
                                                     int rows, int cols) {
  int i = blockIdx.x * 256 + threadIdx.x;
  if (i < rows * cols) {
    int c = i / rows;
    int k = i % rows;
    dst[i] = f2bf(src[k * cols + c]);
  }
}

// ---------------- QKV projection ----------------
// One wave computes a 16x64 tile: A-fragment reused across 4 B-tiles (4 chained WMMAs
// per A load).  Tiles: 256 (M) x 36 (N/64).
__global__ __launch_bounds__(128) void qkv_gemm(const unsigned short* __restrict__ Xh,
                                                const unsigned short* __restrict__ WqkvT,
                                                const float* __restrict__ bqkv,
                                                unsigned short* __restrict__ Qh,
                                                unsigned short* __restrict__ Kh,
                                                unsigned short* __restrict__ Vt) {
  const int lane = threadIdx.x & 31;
  const int w    = threadIdx.x >> 5;
  const int t    = blockIdx.x * 4 + w;
  const int m0   = (t / 36) * 16;
  const int c0   = (t % 36) * 64;
  const int nn   = lane & 15;
  const int g    = lane >> 4;

  const unsigned short* arow = Xh + (size_t)(m0 + nn) * DMODEL;
  const unsigned short* bc[4];
#pragma unroll
  for (int q = 0; q < 4; ++q)
    bc[q] = WqkvT + (size_t)(c0 + 16 * q + nn) * DMODEL + g * 16;

  v8f acc0 = {}, acc1 = {}, acc2 = {}, acc3 = {};
#pragma unroll 2
  for (int ks = 0; ks < DMODEL; ks += 32) {
    const v16bf a = ldA32(arow + ks, g);
    acc0 = wmma_bf16(a, ldB32(bc[0] + ks), acc0);
    acc1 = wmma_bf16(a, ldB32(bc[1] + ks), acc1);
    acc2 = wmma_bf16(a, ldB32(bc[2] + ks), acc2);
    acc3 = wmma_bf16(a, ldB32(bc[3] + ks), acc3);
  }

  v8f accs[4] = {acc0, acc1, acc2, acc3};
#pragma unroll
  for (int q = 0; q < 4; ++q) {
    const int c      = c0 + 16 * q + nn;
    const float bias = bqkv[c];
    const int which  = c % 3;
    const int d      = (c / 3) & 63;
    const int h      = c / 192;
#pragma unroll
    for (int r = 0; r < 8; ++r) {
      const int m = m0 + r + 8 * g;
      const unsigned short v = f2bf(accs[q][r] + bias);
      if (which == 0)      Qh[((size_t)h * NSEQ + m) * DHEAD + d] = v;
      else if (which == 1) Kh[((size_t)h * NSEQ + m) * DHEAD + d] = v;
      else                 Vt[((size_t)h * DHEAD + d) * NSEQ + m] = v;
    }
  }
}

// ---------------- flash attention ----------------
// One workgroup = 4 waves = one head x 64 queries.  Per 32-key block:
//   K tile (contiguous 4KB) -> LDS via per-lane async-to-LDS (ASYNCcnt)
//   V tile (64 x 32, stride 8KB) -> LDS via one TDM tensor_load (TENSORcnt, wave 0)
// both double-buffered so the next block's DMA overlaps this block's WMMAs.
__global__ __launch_bounds__(128) void attn_kernel(const unsigned short* __restrict__ Qh,
                                                   const unsigned short* __restrict__ Kh,
                                                   const unsigned short* __restrict__ Vt,
                                                   unsigned short* __restrict__ AO) {
  __shared__ __align__(16) unsigned short lK[2][32 * DHEAD];  // [key][d]
  __shared__ __align__(16) unsigned short lV[2][DHEAD * 32];  // [d][key]
  __shared__ __align__(16) float          sS[4][16 * 32];
  __shared__ __align__(16) unsigned short sP[4][16 * 32];
  __shared__ __align__(16) float          sA[4][16];

  const int tid  = threadIdx.x;
  const int lane = tid & 31;
  const int w    = tid >> 5;
  const int h    = blockIdx.x / 64;
  const int q0   = (blockIdx.x % 64) * 64 + w * 16;
  const int nn   = lane & 15;
  const int g    = lane >> 4;

  float*          S  = sS[w];
  unsigned short* P  = sP[w];
  float*          AL = sA[w];

  const unsigned short* kbase = Kh + (size_t)h * NSEQ * DHEAD;
  const unsigned short* vbase = Vt + (size_t)h * DHEAD * NSEQ;

  const unsigned lKb[2] = {(unsigned)(size_t)&lK[0][0], (unsigned)(size_t)&lK[1][0]};
  const unsigned lVb[2] = {(unsigned)(size_t)&lV[0][0], (unsigned)(size_t)&lV[1][0]};

  const unsigned short* qrow = Qh + ((size_t)h * NSEQ + q0 + nn) * DHEAD;
  const v16bf aq0 = ldA32(qrow, g);
  const v16bf aq1 = ldA32(qrow + 32, g);

  v8f o0 = {}, o1 = {}, o2 = {}, o3 = {};
  float m_i = -3.0e38f, l_i = 0.f;

  auto stage_k = [&](int kt, int buf) {           // 2 async b128 per thread
#pragma unroll
    for (int j = 0; j < 2; ++j) {
      const int idx = tid + j * 128;              // 0..255 16B chunks of 4KB
      async_copy16(lKb[buf] + idx * 16, kbase, (unsigned)(kt * 128 + idx * 16));
    }
  };
  auto stage_v = [&](int kt, int buf) {           // 1 TDM issue (wave 0 only)
    if (w == 0)
      tdm_load_2d_bf16(lVb[buf], vbase + kt, NSEQ, DHEAD, NSEQ, 32, DHEAD);
  };

  stage_k(0, 0);
  stage_v(0, 0);
  int cur = 0;

  for (int kt = 0; kt < NSEQ; kt += 32) {
    if (kt + 32 < NSEQ) {
      stage_k(kt + 32, cur ^ 1);                  // overlap next block's DMA
      stage_v(kt + 32, cur ^ 1);
      asm volatile("s_wait_asynccnt 0x2" ::: "memory");
      if (w == 0) __builtin_amdgcn_s_wait_tensorcnt(1);
    } else {
      asm volatile("s_wait_asynccnt 0x0" ::: "memory");
      if (w == 0) __builtin_amdgcn_s_wait_tensorcnt(0);
    }
    __syncthreads();

    const unsigned short* Kt = &lK[cur][0];
    const unsigned short* Vl = &lV[cur][0];

    // ---- S = Q @ K^T for this 16x32 key block ----
    v8f s0 = {}, s1 = {};
    s0 = wmma_bf16(aq0, ldB32(Kt + nn * DHEAD + g * 16), s0);
    s0 = wmma_bf16(aq1, ldB32(Kt + nn * DHEAD + 32 + g * 16), s0);
    s1 = wmma_bf16(aq0, ldB32(Kt + (16 + nn) * DHEAD + g * 16), s1);
    s1 = wmma_bf16(aq1, ldB32(Kt + (16 + nn) * DHEAD + 32 + g * 16), s1);

#pragma unroll
    for (int r = 0; r < 8; ++r) {
      S[(r + 8 * g) * 32 + nn]      = s0[r];
      S[(r + 8 * g) * 32 + 16 + nn] = s1[r];
    }
    __syncthreads();

    // ---- online softmax: lane l owns query row l ----
    if (lane < 16) {
      const float* row = S + lane * 32;
      float vbuf[32];
      float rmax = m_i;
#pragma unroll
      for (int j = 0; j < 32; ++j) { vbuf[j] = row[j]; rmax = fmaxf(rmax, vbuf[j]); }
      const float alpha = __expf(m_i - rmax);
      float ssum = 0.f;
      unsigned short* prow = P + lane * 32;
#pragma unroll
      for (int j = 0; j < 32; ++j) {
        const float p = __expf(vbuf[j] - rmax);
        ssum += p;
        prow[j] = f2bf(p);
      }
      l_i = l_i * alpha + ssum;
      m_i = rmax;
      AL[lane] = alpha;
    }
    __syncthreads();

    // ---- rescale O, then O += P @ V ----
#pragma unroll
    for (int r = 0; r < 8; ++r) {
      const float al = AL[r + 8 * g];
      o0[r] *= al; o1[r] *= al; o2[r] *= al; o3[r] *= al;
    }
    const v8bf* pp = (const v8bf*)(P + nn * 32);
    const v16bf ap = cat8(pp[2 * g], pp[2 * g + 1]);
    o0 = wmma_bf16(ap, ldB32(Vl + ( 0 + nn) * 32 + g * 16), o0);
    o1 = wmma_bf16(ap, ldB32(Vl + (16 + nn) * 32 + g * 16), o1);
    o2 = wmma_bf16(ap, ldB32(Vl + (32 + nn) * 32 + g * 16), o2);
    o3 = wmma_bf16(ap, ldB32(Vl + (48 + nn) * 32 + g * 16), o3);
    __syncthreads();
    cur ^= 1;
  }

  // ---- finalize ----
  if (lane < 16) AL[lane] = 1.0f / (l_i * 27.712812921102035f);   // sqrt(768)
  __syncthreads();
#pragma unroll
  for (int r = 0; r < 8; ++r) {
    const float sc = AL[r + 8 * g];
    const int m = q0 + r + 8 * g;
    unsigned short* orow = AO + (size_t)m * DMODEL + h * DHEAD;
    orow[ 0 + nn] = f2bf(o0[r] * sc);
    orow[16 + nn] = f2bf(o1[r] * sc);
    orow[32 + nn] = f2bf(o2[r] * sc);
    orow[48 + nn] = f2bf(o3[r] * sc);
  }
}

// ---------------- output projection (16x64 tile per wave) ----------------
__global__ __launch_bounds__(128) void out_gemm(const unsigned short* __restrict__ AO,
                                                const unsigned short* __restrict__ WoT,
                                                const float* __restrict__ bo,
                                                float* __restrict__ out) {
  const int lane = threadIdx.x & 31;
  const int w    = threadIdx.x >> 5;
  const int t    = blockIdx.x * 4 + w;          // 256 x 12 tiles
  const int m0   = (t / 12) * 16;
  const int c0   = (t % 12) * 64;
  const int nn   = lane & 15;
  const int g    = lane >> 4;

  const unsigned short* arow = AO + (size_t)(m0 + nn) * DMODEL;
  const unsigned short* bc[4];
#pragma unroll
  for (int q = 0; q < 4; ++q)
    bc[q] = WoT + (size_t)(c0 + 16 * q + nn) * DMODEL + g * 16;

  v8f acc0 = {}, acc1 = {}, acc2 = {}, acc3 = {};
#pragma unroll 2
  for (int ks = 0; ks < DMODEL; ks += 32) {
    const v16bf a = ldA32(arow + ks, g);
    acc0 = wmma_bf16(a, ldB32(bc[0] + ks), acc0);
    acc1 = wmma_bf16(a, ldB32(bc[1] + ks), acc1);
    acc2 = wmma_bf16(a, ldB32(bc[2] + ks), acc2);
    acc3 = wmma_bf16(a, ldB32(bc[3] + ks), acc3);
  }

  v8f accs[4] = {acc0, acc1, acc2, acc3};
#pragma unroll
  for (int q = 0; q < 4; ++q) {
    const float bias = bo[c0 + 16 * q + nn];
#pragma unroll
    for (int r = 0; r < 8; ++r)
      out[(size_t)(m0 + r + 8 * g) * DMODEL + c0 + 16 * q + nn] = accs[q][r] + bias;
  }
}

// ---------------- launch ----------------
extern "C" void kernel_launch(void* const* d_in, const int* in_sizes, int n_in,
                              void* d_out, int out_size, void* d_ws, size_t ws_size,
                              hipStream_t stream) {
  const float* x    = (const float*)d_in[0];
  const float* Wqkv = (const float*)d_in[1];
  const float* bqkv = (const float*)d_in[2];
  const float* Wo   = (const float*)d_in[3];
  const float* bo   = (const float*)d_in[4];
  float* out = (float*)d_out;

  unsigned short* Xh    = (unsigned short*)d_ws;
  unsigned short* WqkvT = Xh    + (size_t)NSEQ * DMODEL;
  unsigned short* WoT   = WqkvT + (size_t)D3   * DMODEL;
  unsigned short* Qh    = WoT   + (size_t)DMODEL * DMODEL;
  unsigned short* Kh    = Qh    + (size_t)NHEAD * NSEQ * DHEAD;
  unsigned short* Vt    = Kh    + (size_t)NHEAD * NSEQ * DHEAD;
  unsigned short* AO    = Vt    + (size_t)NHEAD * NSEQ * DHEAD;

  cvt_f32_bf16 <<<(NSEQ * DMODEL) / 256, 256, 0, stream>>>(x, Xh, NSEQ * DMODEL);
  transpose_cvt<<<(DMODEL * D3)   / 256, 256, 0, stream>>>(Wqkv, WqkvT, DMODEL, D3);
  transpose_cvt<<<(DMODEL * DMODEL)/256, 256, 0, stream>>>(Wo, WoT, DMODEL, DMODEL);

  qkv_gemm   <<<(256 * 36) / 4, 128, 0, stream>>>(Xh, WqkvT, bqkv, Qh, Kh, Vt);
  attn_kernel<<<NHEAD * 64, 128, 0, stream>>>(Qh, Kh, Vt, AO);
  out_gemm   <<<(256 * 12) / 4, 128, 0, stream>>>(AO, WoT, bo, out);
}
